// HiSTaR_module_43447889166915
// MI455X (gfx1250) — matrix operand order repaired and checked by reading, not compiled, not run on hardware
//
#include <hip/hip_runtime.h>

// ---------------- problem constants (from reference) ----------------
#define NN    30000
#define EE    480000
#define DIN   3000
#define FH1   64
#define FH2   16
#define GH1   64
#define GH2   16
#define LAT   48
#define NCLU  10
#define NMASK 24000          // int(0.8 * 30000)
#define KPAD  3008           // DIN padded up to multiple of 32
#define BN_RS 0.99950037f    // 1/sqrt(1.001) -- computed at runtime anyway

typedef __attribute__((ext_vector_type(16))) __bf16       v16bf;
typedef __attribute__((ext_vector_type(8)))  float        v8f;
typedef __attribute__((ext_vector_type(4)))  float        f32x4;
typedef __attribute__((ext_vector_type(4)))  unsigned int u32x4;

union AF { v16bf v; __bf16 h[16]; u32x4 q[2]; };

__device__ __forceinline__ float eluf(float v) { return v > 0.f ? v : expm1f(v); }

// ---------------- prep: transpose weights to bf16, K-padded ----------------
// Wt1[n][k]  (64 x 3008)  from W_enc1 (3000 x 64), zeros for k >= 3000
// Wdt[n][k]  (3000 x 64)  from W_dec  (48 x 3000), zeros for k >= 48
__global__ void k_prep(const float* __restrict__ W1, const float* __restrict__ Wd,
                       __bf16* __restrict__ Wt1, __bf16* __restrict__ Wdt)
{
    int t = blockIdx.x * blockDim.x + threadIdx.x;
    const int tot1 = FH1 * KPAD;
    if (t < tot1) {
        int n = t / KPAD, k = t % KPAD;
        Wt1[t] = (__bf16)((k < DIN) ? W1[(long)k * FH1 + n] : 0.0f);
    }
    int t2 = t - tot1;
    if (t2 >= 0 && t2 < DIN * 64) {
        int n = t2 / 64, k = t2 % 64;
        Wdt[t2] = (__bf16)((k < LAT) ? Wd[(long)k * DIN + n] : 0.0f);
    }
}

__global__ void k_setmask(const int* __restrict__ perm, unsigned char* __restrict__ flag)
{
    int t = blockIdx.x * blockDim.x + threadIdx.x;
    if (t < NMASK) flag[perm[t]] = 1;
}

// ---------------- GEMM1: h1 = enc_block(xm @ W_enc1) via bf16 WMMA ----------------
// wave w of block handles M-tile (16 rows), all 64 output cols (4 accumulators).
__global__ void __launch_bounds__(128) k_gemm1(
    const float* __restrict__ x, const float* __restrict__ tok,
    const unsigned char* __restrict__ mflag,
    const __bf16* __restrict__ Wt1,
    const float* __restrict__ b1, const float* __restrict__ g1, const float* __restrict__ bt1,
    float* __restrict__ h1)
{
    const int lane  = threadIdx.x & 31;
    const int wave  = threadIdx.x >> 5;
    const int mtile = blockIdx.x * 4 + wave;
    if (mtile * 16 >= NN) return;                 // wave-uniform

    const int row  = mtile * 16 + (lane & 15);    // A row this lane feeds
    const int base = (lane >> 4) * 8;             // K-offset (A layout)
    const int kb   = (lane >> 4) * 8;             // K-offset (B layout)
    const int col0 = lane & 15;
    const float fm = (mflag[row] != 0) ? 1.0f : 0.0f;  // branchless mask add
    const float* xr = x + (long)row * DIN;

    v8f acc0 = {0.f,0.f,0.f,0.f,0.f,0.f,0.f,0.f};
    v8f acc1 = acc0, acc2 = acc0, acc3 = acc0;

    const __bf16* wcol = Wt1 + (long)col0 * KPAD + kb;

    for (int k0 = 0; k0 + 32 <= DIN; k0 += 32) {
        AF a;
        f32x4 xa = *(const f32x4*)(xr + k0 + base);
        f32x4 xb = *(const f32x4*)(xr + k0 + base + 4);
        f32x4 xc = *(const f32x4*)(xr + k0 + base + 16);
        f32x4 xd = *(const f32x4*)(xr + k0 + base + 20);
        f32x4 ta = *(const f32x4*)(tok + k0 + base);
        f32x4 tb = *(const f32x4*)(tok + k0 + base + 4);
        f32x4 tc = *(const f32x4*)(tok + k0 + base + 16);
        f32x4 td = *(const f32x4*)(tok + k0 + base + 20);
#pragma unroll
        for (int i = 0; i < 4; ++i) {
            a.h[i]      = (__bf16)(xa[i] + fm * ta[i]);
            a.h[4 + i]  = (__bf16)(xb[i] + fm * tb[i]);
            a.h[8 + i]  = (__bf16)(xc[i] + fm * tc[i]);
            a.h[12 + i] = (__bf16)(xd[i] + fm * td[i]);
        }
        const __bf16* wp = wcol + k0;
        AF b0, b1f, b2f, b3f;
        b0.q[0]  = *(const u32x4*)(wp);               b0.q[1]  = *(const u32x4*)(wp + 16);
        b1f.q[0] = *(const u32x4*)(wp + 16 * KPAD);   b1f.q[1] = *(const u32x4*)(wp + 16 * KPAD + 16);
        b2f.q[0] = *(const u32x4*)(wp + 32 * KPAD);   b2f.q[1] = *(const u32x4*)(wp + 32 * KPAD + 16);
        b3f.q[0] = *(const u32x4*)(wp + 48 * KPAD);   b3f.q[1] = *(const u32x4*)(wp + 48 * KPAD + 16);
        acc0 = __builtin_amdgcn_wmma_f32_16x16x32_bf16(false, a.v, false, b0.v,  (short)0, acc0, false, false);
        acc1 = __builtin_amdgcn_wmma_f32_16x16x32_bf16(false, a.v, false, b1f.v, (short)0, acc1, false, false);
        acc2 = __builtin_amdgcn_wmma_f32_16x16x32_bf16(false, a.v, false, b2f.v, (short)0, acc2, false, false);
        acc3 = __builtin_amdgcn_wmma_f32_16x16x32_bf16(false, a.v, false, b3f.v, (short)0, acc3, false, false);
    }
    { // tail: k0 = 2976 (K 2976..3007, valid < 3000; Wt1 zero-padded)
        const int k0 = (DIN / 32) * 32;
        AF a;
#pragma unroll
        for (int i = 0; i < 8; ++i) {
            int k  = k0 + base + i;
            int k2 = k0 + base + 16 + i;
            a.h[i]     = (__bf16)((k  < DIN) ? (xr[k]  + fm * tok[k])  : 0.0f);
            a.h[8 + i] = (__bf16)((k2 < DIN) ? (xr[k2] + fm * tok[k2]) : 0.0f);
        }
        const __bf16* wp = wcol + k0;
        AF b0, b1f, b2f, b3f;
        b0.q[0]  = *(const u32x4*)(wp);               b0.q[1]  = *(const u32x4*)(wp + 16);
        b1f.q[0] = *(const u32x4*)(wp + 16 * KPAD);   b1f.q[1] = *(const u32x4*)(wp + 16 * KPAD + 16);
        b2f.q[0] = *(const u32x4*)(wp + 32 * KPAD);   b2f.q[1] = *(const u32x4*)(wp + 32 * KPAD + 16);
        b3f.q[0] = *(const u32x4*)(wp + 48 * KPAD);   b3f.q[1] = *(const u32x4*)(wp + 48 * KPAD + 16);
        acc0 = __builtin_amdgcn_wmma_f32_16x16x32_bf16(false, a.v, false, b0.v,  (short)0, acc0, false, false);
        acc1 = __builtin_amdgcn_wmma_f32_16x16x32_bf16(false, a.v, false, b1f.v, (short)0, acc1, false, false);
        acc2 = __builtin_amdgcn_wmma_f32_16x16x32_bf16(false, a.v, false, b2f.v, (short)0, acc2, false, false);
        acc3 = __builtin_amdgcn_wmma_f32_16x16x32_bf16(false, a.v, false, b3f.v, (short)0, acc3, false, false);
    }
    // epilogue: enc_block bias/scale/elu
    const float rs = rsqrtf(1.0f + 0.001f);
    const int mloc = mtile * 16 + 8 * (lane >> 4);
#pragma unroll
    for (int nt = 0; nt < 4; ++nt) {
        v8f acc = (nt == 0) ? acc0 : (nt == 1) ? acc1 : (nt == 2) ? acc2 : acc3;
        int j = nt * 16 + (lane & 15);
        float bb = b1[j], gg = g1[j] * rs, be = bt1[j];
#pragma unroll
        for (int r = 0; r < 8; ++r) {
            float v = acc[r] + bb;
            v = v * gg + be;
            h1[(long)(mloc + r) * FH1 + j] = eluf(v);
        }
    }
}

// ---------------- enc2: feat = enc_block(h1 @ W_enc2)  (N x 16) ----------------
__global__ void k_enc2(const float* __restrict__ h1, const float* __restrict__ W2,
                       const float* __restrict__ b2, const float* __restrict__ g2,
                       const float* __restrict__ bt2, float* __restrict__ feat)
{
    int t = blockIdx.x * blockDim.x + threadIdx.x;
    if (t >= NN * FH2) return;
    int i = t >> 4, j = t & 15;
    const float* hr = h1 + (long)i * FH1;
    float acc = 0.f;
#pragma unroll 8
    for (int k = 0; k < FH1; ++k) acc += hr[k] * W2[k * FH2 + j];
    const float rs = rsqrtf(1.0f + 0.001f);
    float v = acc + b2[j];
    v = v * (g2[j] * rs) + bt2[j];
    feat[t] = eluf(v);
}

// ---------------- 16-wide segment-sum spmm (dst must be pre-zeroed) ----------------
__global__ void k_spmm16(const int* __restrict__ erow, const int* __restrict__ ecol,
                         const float* __restrict__ eval, const float* __restrict__ src,
                         float* __restrict__ dst)
{
    int t = blockIdx.x * blockDim.x + threadIdx.x;
    if (t >= EE * 16) return;
    int e = t >> 4, f = t & 15;
    float v = eval[e] * src[(long)ecol[e] * 16 + f];
    atomicAdd(&dst[(long)erow[e] * 16 + f], v);
}

// ---------------- multihop: out = relu((w0*ma + w1*mb) @ Wmh + hin @ Wres + br) ----------------
__global__ void k_multihop(const float* __restrict__ ma, const float* __restrict__ mb,
                           const float* __restrict__ hin, const float* __restrict__ Wmh,
                           const float* __restrict__ hop, const float* __restrict__ Wres,
                           const float* __restrict__ br, float* __restrict__ out)
{
    int t = blockIdx.x * blockDim.x + threadIdx.x;
    if (t >= NN * 64) return;
    int i = t >> 6, n = t & 63;
    float h0 = hop[0], h1 = hop[1];
    float mx = fmaxf(h0, h1);
    float e0 = expf(h0 - mx), e1 = expf(h1 - mx);
    float inv = 1.0f / (e0 + e1);
    float w0 = e0 * inv, w1 = e1 * inv;
    const float* mar = ma + (long)i * 16;
    const float* mbr = mb + (long)i * 16;
    const float* hr  = hin + (long)i * 16;
    float acc = br[n];
#pragma unroll
    for (int k = 0; k < 16; ++k)
        acc += (w0 * mar[k] + w1 * mbr[k]) * Wmh[k * 64 + n] + hr[k] * Wres[k * 64 + n];
    out[t] = fmaxf(acc, 0.f);
}

// ---------------- dual projection: p = hid @ Wmu, pl = hid @ Wlv ----------------
__global__ void k_proj(const float* __restrict__ hid, const float* __restrict__ Wmu,
                       const float* __restrict__ Wlv, float* __restrict__ p, float* __restrict__ pl)
{
    int t = blockIdx.x * blockDim.x + threadIdx.x;
    if (t >= NN * 16) return;
    int i = t >> 4, j = t & 15;
    const float* hr = hid + (long)i * 64;
    float a = 0.f, b = 0.f;
#pragma unroll 8
    for (int k = 0; k < 64; ++k) {
        float h = hr[k];
        a += h * Wmu[k * 16 + j];
        b += h * Wlv[k * 16 + j];
    }
    p[t] = a; pl[t] = b;
}

// ---------------- pack z rows and zs = spmm(z) as bf16 (K padded to 64) ----------------
__global__ void k_pack(const float* __restrict__ feat, const float* __restrict__ mu1,
                       const float* __restrict__ mu2, const float* __restrict__ m1a,
                       const float* __restrict__ m2a, const float* __restrict__ mmu2,
                       float* __restrict__ z, __bf16* __restrict__ zs)
{
    int i = blockIdx.x * blockDim.x + threadIdx.x;
    if (i >= NN) return;
#pragma unroll
    for (int k = 0; k < 16; ++k) {
        z[(long)i * LAT + k]      = feat[(long)i * 16 + k];
        z[(long)i * LAT + 16 + k] = mu1[(long)i * 16 + k];
        z[(long)i * LAT + 32 + k] = mu2[(long)i * 16 + k];
        zs[(long)i * 64 + k]      = (__bf16)m1a[(long)i * 16 + k];
        zs[(long)i * 64 + 16 + k] = (__bf16)m2a[(long)i * 16 + k];
        zs[(long)i * 64 + 32 + k] = (__bf16)mmu2[(long)i * 16 + k];
        zs[(long)i * 64 + 48 + k] = (__bf16)0.0f;
    }
}

// ---------------- decoder GEMM: de = zs @ W_dec  (N x 3000) via bf16 WMMA ----------------
#define NTILES_DEC 188
__global__ void __launch_bounds__(128) k_dec(const __bf16* __restrict__ zs,
                                             const __bf16* __restrict__ Wdt,
                                             float* __restrict__ de)
{
    const int lane  = threadIdx.x & 31;
    const int wave  = threadIdx.x >> 5;
    const int ntile = blockIdx.x * 4 + wave;
    if (ntile >= NTILES_DEC) return;              // wave-uniform
    const int mtile = blockIdx.y;
    const int row  = mtile * 16 + (lane & 15);
    const int base = (lane >> 4) * 8;
    const int kb   = (lane >> 4) * 8;
    const int ncol = ntile * 16 + (lane & 15);
    const int ncl  = (ncol < DIN) ? ncol : (DIN - 1);  // branchless clamp for loads
    const __bf16* ar = zs + (long)row * 64;
    const __bf16* wp = Wdt + (long)ncl * 64 + kb;
    v8f acc = {0.f,0.f,0.f,0.f,0.f,0.f,0.f,0.f};
#pragma unroll
    for (int k0 = 0; k0 < 64; k0 += 32) {
        AF a, b;
        a.q[0] = *(const u32x4*)(ar + k0 + base);
        a.q[1] = *(const u32x4*)(ar + k0 + base + 16);
        b.q[0] = *(const u32x4*)(wp + k0);
        b.q[1] = *(const u32x4*)(wp + k0 + 16);
        acc = __builtin_amdgcn_wmma_f32_16x16x32_bf16(false, a.v, false, b.v, (short)0, acc, false, false);
    }
    const int mloc = mtile * 16 + 8 * (lane >> 4);
    if (ncol < DIN) {
#pragma unroll
        for (int r = 0; r < 8; ++r) de[(long)(mloc + r) * DIN + ncol] = acc[r];
    }
}

// ---------------- soft cluster assignment q (alpha = 1) ----------------
__global__ void k_q(const float* __restrict__ z, const float* __restrict__ cluster,
                    float* __restrict__ q)
{
    int i = blockIdx.x * blockDim.x + threadIdx.x;
    if (i >= NN) return;
    const float* zr = z + (long)i * LAT;
    float qs[NCLU];
    float s = 0.f;
#pragma unroll
    for (int c = 0; c < NCLU; ++c) {
        float d2 = 0.f;
        const float* cl = cluster + c * LAT;
#pragma unroll 8
        for (int k = 0; k < LAT; ++k) { float d = zr[k] - cl[k]; d2 += d * d; }
        float qq = 1.0f / (1.0f + d2);   // alpha=1, exponent (alpha+1)/2 = 1
        qs[c] = qq; s += qq;
    }
    float inv = 1.0f / s;
#pragma unroll
    for (int c = 0; c < NCLU; ++c) q[(long)i * NCLU + c] = qs[c] * inv;
}

// ---------------- cosine(mu1, mu2) sum -> acc[0] ----------------
__global__ void k_cos(const float* __restrict__ mu1, const float* __restrict__ mu2,
                      float* __restrict__ acc)
{
    int i = blockIdx.x * blockDim.x + threadIdx.x;
    if (i >= NN) return;
    float d = 0.f, a = 0.f, b = 0.f;
#pragma unroll
    for (int k = 0; k < 16; ++k) {
        float xv = mu1[(long)i * 16 + k], yv = mu2[(long)i * 16 + k];
        d += xv * yv; a += xv * xv; b += yv * yv;
    }
    float c = d / (fmaxf(sqrtf(a), 1e-8f) * fmaxf(sqrtf(b), 1e-8f));
    atomicAdd(acc, c);
}

// ---------------- masked cosine-cubed recon sum -> acc[1] ----------------
__global__ void __launch_bounds__(256) k_recon(const float* __restrict__ de,
                                               const float* __restrict__ x,
                                               const float* __restrict__ tok,
                                               const int* __restrict__ perm,
                                               float* __restrict__ acc)
{
    __shared__ float sd[256], sr[256], si[256];
    const int m = perm[blockIdx.x];
    const float* dr = de + (long)m * DIN;
    const float* xr = x + (long)m * DIN;
    float d = 0.f, r2 = 0.f, i2 = 0.f;
    for (int k = threadIdx.x; k < DIN; k += 256) {
        float r = dr[k];
        float xi = xr[k] + tok[k];   // masked rows by construction
        d += r * xi; r2 += r * r; i2 += xi * xi;
    }
    sd[threadIdx.x] = d; sr[threadIdx.x] = r2; si[threadIdx.x] = i2;
    __syncthreads();
    for (int s = 128; s > 0; s >>= 1) {
        if (threadIdx.x < s) {
            sd[threadIdx.x] += sd[threadIdx.x + s];
            sr[threadIdx.x] += sr[threadIdx.x + s];
            si[threadIdx.x] += si[threadIdx.x + s];
        }
        __syncthreads();
    }
    if (threadIdx.x == 0) {
        float nr = fmaxf(sqrtf(sr[0]), 1e-12f);
        float ni = fmaxf(sqrtf(si[0]), 1e-12f);
        float c = 1.0f - sd[0] / (nr * ni);
        atomicAdd(acc + 1, c * c * c);
    }
}

// ---------------- final scalar loss ----------------
__device__ __forceinline__ float hop_kl(float h0, float h1)
{
    float mx = fmaxf(h0, h1);
    float e0 = expf(h0 - mx), e1 = expf(h1 - mx);
    float s = e0 + e1;
    float p0 = e0 / s, p1 = e1 / s;
    const float t0 = 0.7f, t1 = 0.3f;
    return t0 * (logf(t0) - logf(p0)) + t1 * (logf(t1) - logf(p1));
}

__global__ void k_final(const float* __restrict__ acc, const float* __restrict__ hop1,
                        const float* __restrict__ hop2, float* __restrict__ loss)
{
    float recon = acc[1] / (float)NMASK;
    float sim   = -acc[0] / (float)NN;
    *loss = recon + 0.3f * sim + 0.15f * (hop_kl(hop1[0], hop1[1]) + hop_kl(hop2[0], hop2[1]));
}

// ==================== host launch ====================
extern "C" void kernel_launch(void* const* d_in, const int* in_sizes, int n_in,
                              void* d_out, int out_size, void* d_ws, size_t ws_size,
                              hipStream_t stream)
{
    const float* x     = (const float*)d_in[0];
    const int*   erow  = (const int*)  d_in[1];
    const int*   ecol  = (const int*)  d_in[2];
    const float* eval  = (const float*)d_in[3];
    const int*   perm  = (const int*)  d_in[4];
    const float* W1    = (const float*)d_in[5];
    const float* b1    = (const float*)d_in[6];
    const float* g1    = (const float*)d_in[7];
    const float* bt1   = (const float*)d_in[8];
    const float* W2    = (const float*)d_in[9];
    const float* b2    = (const float*)d_in[10];
    const float* g2    = (const float*)d_in[11];
    const float* bt2   = (const float*)d_in[12];
    const float* Wmh1  = (const float*)d_in[13];
    const float* hop1  = (const float*)d_in[14];
    const float* Wres1 = (const float*)d_in[15];
    const float* bres1 = (const float*)d_in[16];
    const float* Wmu1  = (const float*)d_in[17];
    const float* Wlv1  = (const float*)d_in[18];
    const float* Wmh2  = (const float*)d_in[19];
    const float* hop2  = (const float*)d_in[20];
    const float* Wres2 = (const float*)d_in[21];
    const float* bres2 = (const float*)d_in[22];
    const float* Wmu2  = (const float*)d_in[23];
    const float* Wlv2  = (const float*)d_in[24];
    const float* Wdec  = (const float*)d_in[25];
    const float* clus  = (const float*)d_in[26];
    const float* tok   = (const float*)d_in[27];

    // ----- d_out layout (return-order concatenation) -----
    float* out    = (float*)d_out;
    float* o_z    = out;                                   // N x 48
    float* o_mu1  = o_z   + (size_t)NN * LAT;              // N x 16
    float* o_mu2  = o_mu1 + (size_t)NN * 16;
    float* o_lv1  = o_mu2 + (size_t)NN * 16;
    float* o_lv2  = o_lv1 + (size_t)NN * 16;
    float* o_de   = o_lv2 + (size_t)NN * 16;               // N x 3000
    float* o_q    = o_de  + (size_t)NN * DIN;              // N x 10
    float* o_feat = o_q   + (size_t)NN * NCLU;             // N x 16
    float* o_loss = o_feat + (size_t)NN * 16;              // 1

    // ----- workspace carve (~45 MB) -----
    char*  wsb = (char*)d_ws;
    size_t off = 0;
    auto carve = [&](size_t bytes) -> void* {
        void* p = wsb + off;
        off = (off + bytes + 255) & ~(size_t)255;
        return p;
    };
    __bf16*        Wt1   = (__bf16*)carve((size_t)FH1 * KPAD * 2);
    __bf16*        Wdt   = (__bf16*)carve((size_t)DIN * 64 * 2);
    unsigned char* mflag = (unsigned char*)carve(NN);
    float*         h1    = (float*)carve((size_t)NN * FH1 * 4);
    float*         m1a   = (float*)carve((size_t)NN * 16 * 4);
    float*         m1b   = (float*)carve((size_t)NN * 16 * 4);
    float*         m2a   = (float*)carve((size_t)NN * 16 * 4);
    float*         m2b   = (float*)carve((size_t)NN * 16 * 4);
    float*         mmu2  = (float*)carve((size_t)NN * 16 * 4);
    float*         hid1  = (float*)carve((size_t)NN * 64 * 4);
    float*         hid2  = (float*)carve((size_t)NN * 64 * 4);
    float*         p1    = (float*)carve((size_t)NN * 16 * 4);
    float*         pl1   = (float*)carve((size_t)NN * 16 * 4);
    float*         p2    = (float*)carve((size_t)NN * 16 * 4);
    float*         pl2   = (float*)carve((size_t)NN * 16 * 4);
    __bf16*        zs    = (__bf16*)carve((size_t)NN * 64 * 2);
    float*         accb  = (float*)carve(256);
    (void)ws_size; (void)in_sizes; (void)n_in; (void)out_size;

    const size_t b16 = (size_t)NN * 16 * 4;
    // zero atomic-accumulation targets (fresh every call -> deterministic work)
    hipMemsetAsync(mflag, 0, NN, stream);
    hipMemsetAsync(m1a, 0, b16, stream);  hipMemsetAsync(m1b, 0, b16, stream);
    hipMemsetAsync(m2a, 0, b16, stream);  hipMemsetAsync(m2b, 0, b16, stream);
    hipMemsetAsync(mmu2, 0, b16, stream); hipMemsetAsync(accb, 0, 256, stream);
    hipMemsetAsync(o_mu1, 0, b16, stream); hipMemsetAsync(o_mu2, 0, b16, stream);
    hipMemsetAsync(o_lv1, 0, b16, stream); hipMemsetAsync(o_lv2, 0, b16, stream);

    // prep
    {
        int tot = FH1 * KPAD + DIN * 64;
        k_prep<<<(tot + 255) / 256, 256, 0, stream>>>(W1, Wdec, Wt1, Wdt);
        k_setmask<<<(NMASK + 255) / 256, 256, 0, stream>>>(perm, mflag);
    }

    // encoder
    k_gemm1<<<dim3((NN / 16 + 3) / 4), dim3(128), 0, stream>>>(x, tok, mflag, Wt1, b1, g1, bt1, h1);
    k_enc2<<<(NN * 16 + 255) / 256, 256, 0, stream>>>(h1, W2, b2, g2, bt2, o_feat);

    const int spmmGrid = (EE * 16 + 255) / 256;
    // multihop 1  (spmm(h@W) == spmm(h)@W)
    k_spmm16<<<spmmGrid, 256, 0, stream>>>(erow, ecol, eval, o_feat, m1a);
    k_spmm16<<<spmmGrid, 256, 0, stream>>>(erow, ecol, eval, m1a, m1b);
    k_multihop<<<(NN * 64 + 255) / 256, 256, 0, stream>>>(m1a, m1b, o_feat, Wmh1, hop1, Wres1, bres1, hid1);
    k_proj<<<(NN * 16 + 255) / 256, 256, 0, stream>>>(hid1, Wmu1, Wlv1, p1, pl1);
    k_spmm16<<<spmmGrid, 256, 0, stream>>>(erow, ecol, eval, p1,  o_mu1);
    k_spmm16<<<spmmGrid, 256, 0, stream>>>(erow, ecol, eval, pl1, o_lv1);

    // multihop 2
    k_spmm16<<<spmmGrid, 256, 0, stream>>>(erow, ecol, eval, o_mu1, m2a);
    k_spmm16<<<spmmGrid, 256, 0, stream>>>(erow, ecol, eval, m2a, m2b);
    k_multihop<<<(NN * 64 + 255) / 256, 256, 0, stream>>>(m2a, m2b, o_mu1, Wmh2, hop2, Wres2, bres2, hid2);
    k_proj<<<(NN * 16 + 255) / 256, 256, 0, stream>>>(hid2, Wmu2, Wlv2, p2, pl2);
    k_spmm16<<<spmmGrid, 256, 0, stream>>>(erow, ecol, eval, p2,  o_mu2);
    k_spmm16<<<spmmGrid, 256, 0, stream>>>(erow, ecol, eval, pl2, o_lv2);
    k_spmm16<<<spmmGrid, 256, 0, stream>>>(erow, ecol, eval, o_mu2, mmu2);

    // z, zs = spmm(z) (reusing m1a = spmm(feat), m2a = spmm(mu1))
    k_pack<<<(NN + 255) / 256, 256, 0, stream>>>(o_feat, o_mu1, o_mu2, m1a, m2a, mmu2, o_z, zs);

    // decoder: de = spmm(z) @ W_dec
    k_dec<<<dim3((NTILES_DEC + 3) / 4, NN / 16), dim3(128), 0, stream>>>(zs, Wdt, o_de);

    // losses + q
    k_q<<<(NN + 255) / 256, 256, 0, stream>>>(o_z, clus, o_q);
    k_cos<<<(NN + 255) / 256, 256, 0, stream>>>(o_mu1, o_mu2, accb);
    k_recon<<<NMASK, 256, 0, stream>>>(o_de, x, tok, perm, accb);
    k_final<<<1, 1, 0, stream>>>(accb, hop1, hop2, o_loss);
}
